// BFConv2d_57406532878907
// MI455X (gfx1250) — matrix operand order repaired and checked by reading, not compiled, not run on hardware
//
#include <hip/hip_runtime.h>

typedef _Float16 v16h __attribute__((ext_vector_type(16)));
typedef _Float16 h8   __attribute__((ext_vector_type(8)));
typedef _Float16 h4   __attribute__((ext_vector_type(4)));
typedef float    v8f  __attribute__((ext_vector_type(8)));

#define GROUP 36
#define QBITS 8

#define BATCH 32
#define CIN   128
#define COUT  128
#define HH    112
#define WW    112
#define HWSZ  (HH * WW)          // 12544
#define PW    114                // padded width
#define PPLANE (PW * PW)         // 12996
#define KTOT  (CIN * 9)          // 1152
#define NIT   (KTOT / 32)        // 36 k-steps
#define NTOT  (BATCH * HWSZ)     // 401408
#define NTILE 128
#define LDA   40                 // LDS row stride (halves), 16B-aligned rows
#define LDB   40

#if __has_builtin(__builtin_amdgcn_tensor_load_to_lds) && __has_builtin(__builtin_amdgcn_s_wait_tensorcnt)
#define HAVE_TDM 1
typedef unsigned int u32x4 __attribute__((ext_vector_type(4)));
typedef int          i32x4 __attribute__((ext_vector_type(4)));
typedef int          i32x8 __attribute__((ext_vector_type(8)));

__device__ __forceinline__ unsigned lds_byte_off(const void* p) {
  // generic LDS pointer: low 32 bits are the LDS byte offset (aperture map)
  return (unsigned)(unsigned long long)(uintptr_t)p;
}

// TDM: 2-D weight tile, 128 rows x 32 halves, global row stride 1152 halves.
// pad_enable: +4 DWORDs after every 16 DWORDs stored -> LDS row stride 40 halves.
__device__ __forceinline__ void tdm_load_A(const _Float16* gsrc, unsigned lds_off) {
  unsigned long long ga = (unsigned long long)(uintptr_t)gsrc;
  u32x4 g0 = { 1u,                                   // count=1 (valid descriptor)
               lds_off,                               // lds_addr (bytes)
               (unsigned)(ga & 0xFFFFFFFFu),          // global_addr[31:0]
               (unsigned)((ga >> 32) & 0x1FFFFFFu)    // global_addr[56:32]
                 | 0x80000000u };                     // type=2 ("image")
  i32x8 g1 = { (int)((1u << 16)      // data_size = 1 -> 2 bytes
                   | (1u << 20)      // pad_enable
                   | (3u << 22)      // pad_interval: 16 DWORDs
                   | (3u << 25)),    // pad_amount: 4 DWORDs
               (int)((KTOT & 0xFFFF) << 16),   // tensor_dim0 = 1152 (bits 79:48)
               (int)(COUT << 16),              // tensor_dim1 = 128  (bits 111:80)
               (int)(32 << 16),                // tile_dim0 = 32     (bits 127:112)
               (int)COUT,                      // tile_dim1 = 128    (bits 143:128)
               (int)KTOT,                      // tensor_dim0_stride = 1152
               0, 0 };
  i32x4 z4 = { 0, 0, 0, 0 };
  i32x8 z8 = { 0, 0, 0, 0, 0, 0, 0, 0 };
  __builtin_amdgcn_tensor_load_to_lds(g0, g1, z4, z4, z8, 0);
}
#endif

// ---------------------------------------------------------------------------
// Zero-fill padded fp16 image buffer (borders must be 0 for the conv gather).
// ---------------------------------------------------------------------------
__global__ void zero_fill_kernel(uint4* __restrict__ p, int n4) {
  int i = blockIdx.x * blockDim.x + threadIdx.x;
  if (i < n4) p[i] = make_uint4(0u, 0u, 0u, 0u);
}

// ---------------------------------------------------------------------------
// BFP quantize weights (flat OIHW group order) and scatter into K-reordered
// layout wq2[co][k'] with k' = (kh*3+kw)*128 + ci, so a 32-wide k-step has a
// constant (kh,kw) and linear ci -> branch/div-free conv gather.
// ---------------------------------------------------------------------------
__global__ void bfp_quant_w_kernel(const float* __restrict__ in,
                                   _Float16* __restrict__ out,
                                   int ngroups) {
  int g = blockIdx.x * blockDim.x + threadIdx.x;
  if (g >= ngroups) return;
  int base = g * GROUP;                       // 144*g bytes -> 16B aligned
  float v[GROUP];
  const float4* p = (const float4*)(in + base);
#pragma unroll
  for (int i = 0; i < 9; ++i) {
    float4 t = p[i];
    v[4 * i + 0] = t.x; v[4 * i + 1] = t.y;
    v[4 * i + 2] = t.z; v[4 * i + 3] = t.w;
  }
  float m = 0.0f;
#pragma unroll
  for (int i = 0; i < GROUP; ++i) m = fmaxf(m, fabsf(v[i]));
  _Float16 q[GROUP];
  if (m > 0.0f) {
    int e = (int)((__float_as_uint(m) >> 23) & 0xFF) - 127;
    float step  = __uint_as_float((unsigned)(e - (QBITS - 1) + 127) << 23);
    float rstep = __uint_as_float((unsigned)(127 - (e - (QBITS - 1))) << 23);
#pragma unroll
    for (int i = 0; i < GROUP; ++i) {
      float t = rintf(v[i] * rstep);
      t = fminf(fmaxf(t, -128.0f), 127.0f);
      q[i] = (_Float16)(t * step);
    }
  } else {
#pragma unroll
    for (int i = 0; i < GROUP; ++i) q[i] = (_Float16)0.0f;
  }
#pragma unroll
  for (int i = 0; i < GROUP; ++i) {
    int f   = base + i;
    int co  = f / KTOT;
    int rem = f - co * KTOT;
    int ci  = rem / 9;
    int r   = rem - ci * 9;
    out[co * KTOT + r * CIN + ci] = q[i];
  }
}

// ---------------------------------------------------------------------------
// BFP quantize x and scatter into zero-padded [B][Cin][114][114] fp16 layout.
// Group boundaries follow the ORIGINAL flat order (matches reference).
// ---------------------------------------------------------------------------
__global__ void bfp_quant_pad_kernel(const float* __restrict__ in,
                                     _Float16* __restrict__ outp,
                                     long long n, int ngroups) {
  int g = blockIdx.x * blockDim.x + threadIdx.x;
  if (g >= ngroups) return;
  long long base = (long long)g * GROUP;
  bool full = (base + GROUP) <= n;
  float v[GROUP];
  if (full) {
    const float4* p = (const float4*)(in + base);
#pragma unroll
    for (int i = 0; i < 9; ++i) {
      float4 t = p[i];
      v[4 * i + 0] = t.x; v[4 * i + 1] = t.y;
      v[4 * i + 2] = t.z; v[4 * i + 3] = t.w;
    }
  } else {
#pragma unroll
    for (int i = 0; i < GROUP; ++i) v[i] = (base + i < n) ? in[base + i] : 0.0f;
  }
  float m = 0.0f;
#pragma unroll
  for (int i = 0; i < GROUP; ++i) m = fmaxf(m, fabsf(v[i]));
  _Float16 q[GROUP];
  if (m > 0.0f) {
    int e = (int)((__float_as_uint(m) >> 23) & 0xFF) - 127;
    float step  = __uint_as_float((unsigned)(e - (QBITS - 1) + 127) << 23);
    float rstep = __uint_as_float((unsigned)(127 - (e - (QBITS - 1))) << 23);
#pragma unroll
    for (int i = 0; i < GROUP; ++i) {
      float t = rintf(v[i] * rstep);
      t = fminf(fmaxf(t, -128.0f), 127.0f);
      q[i] = (_Float16)(t * step);
    }
  } else {
#pragma unroll
    for (int i = 0; i < GROUP; ++i) q[i] = (_Float16)0.0f;
  }
  // scatter into padded layout with incremental offset updates
  int idx = (int)base;
  int t1 = idx / WW;            // (b*CIN + c)*HH + h
  int w  = idx - t1 * WW;
  int t2 = t1 / HH;             // b*CIN + c
  int h  = t1 - t2 * HH;
  int off = t2 * PPLANE + (h + 1) * PW + (w + 1);
#pragma unroll
  for (int i = 0; i < GROUP; ++i) {
    if (base + i < n) outp[off] = q[i];
    ++w; ++off;
    if (w == WW) {
      w = 0; off += 2; ++h;
      if (h == HH) { h = 0; off += 2 * PW; }
    }
  }
}

// ---------------------------------------------------------------------------
// Implicit-GEMM conv (3x3, pad 1) on v_wmma_f32_16x16x32_f16.
// M = Cout = 128, N = B*H*W = 401408, K' = 9*128 (r-major, ci-minor).
// Block: 256 thr = 8 waves, tile 128(M) x 128(N); wave = 16 M-rows x 8 N-tiles
// -> 8 WMMAs per k-step, 36 k-steps. Double-buffered LDS, TDM weight staging,
// register-pipelined B gather with immediate-offset strided loads.
// ---------------------------------------------------------------------------
__global__ __launch_bounds__(256)
void bfp_conv_wmma_kernel(const _Float16* __restrict__ xq,  // padded image
                          const _Float16* __restrict__ wq,  // [Cout][k'=1152]
                          const float* __restrict__ bias,
                          float* __restrict__ out) {
  __shared__ _Float16 Alds[2][COUT][LDA];
  __shared__ _Float16 Blds[2][NTILE][LDB];

  const int tid  = threadIdx.x;
  const int lane = tid & 31;
  const int wid  = tid >> 5;      // 0..7
  const int l    = lane & 15;
  const int hi   = lane >> 4;
  const int m0   = wid << 4;
  const int nbase = blockIdx.x * NTILE;   // NTOT % NTILE == 0

  v8f acc[8];
#pragma unroll
  for (int t = 0; t < 8; ++t) acc[t] = 0;

  // cooperative-load roles (B tile: 128 n x 32 k'; 16 elements per thread)
  const int bn  = tid & 127;            // local n
  const int bkc = (tid >> 7) << 4;      // k' chunk: 0 or 16

  const int n_g   = nbase + bn;
  const int b_g   = n_g / HWSZ;
  const int rem_g = n_g - b_g * HWSZ;
  const int h_g   = rem_g / WW;
  const int w_g   = rem_g - h_g * WW;
  // padded gather base: hin_pad = h_g + kh, win_pad = w_g + kw (pad+1, conv-1)
  const int base_pix = b_g * CIN * PPLANE + h_g * PW + w_g;

  // B gather for k-step base kb: r = kb>>7 (uniform), ci = (kb&127)+bkc+j
  // addr = base_pix + kh*PW + kw + ci*PPLANE; j*PPLANE folds into load offset.
#define STAGE_B_REGS(kb, breg)                                              \
  {                                                                          \
    int r_  = (kb) >> 7;                                                     \
    int kh_ = r_ / 3;                                                        \
    int kw_ = r_ - kh_ * 3;                                                  \
    const _Float16* p_ =                                                     \
        xq + base_pix + kh_ * PW + kw_ + (((kb) & 127) + bkc) * PPLANE;      \
    _Pragma("unroll")                                                        \
    for (int j = 0; j < 16; ++j) (breg)[j] = p_[j * PPLANE];                 \
  }

#define STORE_B_LDS(buf, breg)                                               \
  {                                                                          \
    _Pragma("unroll")                                                        \
    for (int c = 0; c < 4; ++c) {                                            \
      h4 t_;                                                                 \
      t_[0] = (breg)[4 * c + 0]; t_[1] = (breg)[4 * c + 1];                  \
      t_[2] = (breg)[4 * c + 2]; t_[3] = (breg)[4 * c + 3];                  \
      *(h4*)&Blds[(buf)][bn][bkc + 4 * c] = t_;                              \
    }                                                                        \
  }

  // ---------------- prologue: stage k-step 0 into buffer 0 ----------------
#if defined(HAVE_TDM)
  if (wid == 0) tdm_load_A(wq, lds_byte_off(&Alds[0][0][0]));
#else
  {
    const int ar = tid & 127;
    const int ac = (tid >> 7) << 4;
    const h8* s = (const h8*)(wq + ar * KTOT + ac);
    *(h8*)&Alds[0][ar][ac]     = s[0];
    *(h8*)&Alds[0][ar][ac + 8] = s[1];
  }
#endif
  {
    _Float16 breg0[16];
    STAGE_B_REGS(0, breg0)
    STORE_B_LDS(0, breg0)
  }
#if defined(HAVE_TDM)
  if (wid == 0) __builtin_amdgcn_s_wait_tensorcnt(0);
#endif
  __syncthreads();

  // ---------------- main loop: one barrier per k-step ----------------
  for (int it = 0; it < NIT; ++it) {
    const int cur  = it & 1;
    const int nxt  = cur ^ 1;
    const bool more = (it + 1 < NIT);
    const int kb_n = (it + 1) * 32;

    // issue next tile's global traffic before computing
    _Float16 breg[16];
#if !defined(HAVE_TDM)
    h8 areg0, areg1;
#endif
    if (more) {
#if defined(HAVE_TDM)
      if (wid == 0) tdm_load_A(wq + kb_n, lds_byte_off(&Alds[nxt][0][0]));
#else
      const int ar = tid & 127;
      const int ac = (tid >> 7) << 4;
      const h8* s = (const h8*)(wq + ar * KTOT + kb_n + ac);
      areg0 = s[0];
      areg1 = s[1];
#endif
      STAGE_B_REGS(kb_n, breg)
    }

    // ---- A fragment: lanes 0-15 K0-7 & K16-23, lanes 16-31 K8-15 & K24-31
    v16h a;
    {
      h8 alo = *(const h8*)&Alds[cur][m0 + l][hi * 8];
      h8 ahi = *(const h8*)&Alds[cur][m0 + l][16 + hi * 8];
#pragma unroll
      for (int j = 0; j < 8; ++j) { a[j] = alo[j]; a[j + 8] = ahi[j]; }
    }

    // ---- 8 B fragments + 8 WMMAs: N = lane%16, lanes 0-15 K0-7 & K16-23, lanes 16-31 K8-15 & K24-31
#pragma unroll
    for (int t = 0; t < 8; ++t) {
      const _Float16* bp = &Blds[cur][t * 16 + l][hi * 8];
      h8 blo = *(const h8*)bp;
      h8 bhi = *(const h8*)(bp + 16);
      v16h b;
#pragma unroll
      for (int j = 0; j < 8; ++j) { b[j] = blo[j]; b[j + 8] = bhi[j]; }
      acc[t] = __builtin_amdgcn_wmma_f32_16x16x32_f16(
          false, a, false, b, (short)0, acc[t], false, false);
    }

    if (more) {
#if !defined(HAVE_TDM)
      const int ar = tid & 127;
      const int ac = (tid >> 7) << 4;
      *(h8*)&Alds[nxt][ar][ac]     = areg0;
      *(h8*)&Alds[nxt][ar][ac + 8] = areg1;
#endif
      STORE_B_LDS(nxt, breg)
#if defined(HAVE_TDM)
      if (wid == 0) __builtin_amdgcn_s_wait_tensorcnt(0);
#endif
      __syncthreads();
    }
  }

  // ---- epilogue: C layout VGPR r -> M = r + 8*hi, N = lane%16; add bias ----
#pragma unroll
  for (int t = 0; t < 8; ++t) {
    int n   = nbase + t * 16 + l;
    int b   = n / HWSZ;
    int rem = n - b * HWSZ;
#pragma unroll
    for (int r = 0; r < 8; ++r) {
      int m = m0 + r + hi * 8;
      out[(b * COUT + m) * HWSZ + rem] = acc[t][r] + bias[m];
    }
  }
#undef STAGE_B_REGS
#undef STORE_B_LDS
}

// ---------------------------------------------------------------------------
extern "C" void kernel_launch(void* const* d_in, const int* in_sizes, int n_in,
                              void* d_out, int out_size, void* d_ws, size_t ws_size,
                              hipStream_t stream) {
  (void)in_sizes; (void)n_in; (void)out_size; (void)ws_size;

  const float* x    = (const float*)d_in[0];  // (32,128,112,112)
  const float* w    = (const float*)d_in[1];  // (128,128,3,3)
  const float* bias = (const float*)d_in[2];  // (128,)
  float* out = (float*)d_out;

  const long long nx     = (long long)BATCH * CIN * HH * WW;   // 51,380,224
  const long long nx_pad = (long long)BATCH * CIN * PPLANE;    // 53,231,616
  const long long nw     = (long long)COUT * CIN * 9;          // 147,456

  _Float16* xq_pad = (_Float16*)d_ws;
  _Float16* wq     = xq_pad + nx_pad;   // total ws use ~101.8 MiB

  const int n4 = (int)(nx_pad * 2 / 16);            // uint4 count (exact)
  const int gx = (int)((nx + GROUP - 1) / GROUP);   // 1,427,229 (last partial)
  const int gw = (int)(nw / GROUP);                 // 4,096 (exact)

  zero_fill_kernel<<<(n4 + 255) / 256, 256, 0, stream>>>((uint4*)xq_pad, n4);
  bfp_quant_pad_kernel<<<(gx + 255) / 256, 256, 0, stream>>>(x, xq_pad, nx, gx);
  bfp_quant_w_kernel<<<(gw + 255) / 256, 256, 0, stream>>>(w, wq, gw);
  bfp_conv_wmma_kernel<<<NTOT / NTILE, 256, 0, stream>>>(xq_pad, wq, bias, out);
}
